// PersonalizedAttention_52432960749921
// MI455X (gfx1250) — matrix operand order repaired and checked by reading, not compiled, not run on hardware
//
#include <hip/hip_runtime.h>
#include <hip/hip_bf16.h>

#define BATCH 2048
#define SEQ   200
#define KD    400   // QUERY_DIM == KEY_DIM
#define HD    200   // HIDDEN_DIM

#define SM    208   // padded SEQ   (13 * 16)
#define KP    424   // keys LDS stride in bf16 elems (>=416=13*32; 16B aligned rows, conflict-free)
#define KPW   424   // wk LDS stride in bf16 elems
#define NH0   112   // first  N-half: 7 tiles of 16
#define NH1   96    // second N-half: 6 tiles of 16
#define HP    208   // padded hidden
#define NWAVES 8

typedef __attribute__((ext_vector_type(16))) __bf16 v16bf;
typedef __attribute__((ext_vector_type(8)))  float  v8f;

union Frag { unsigned int u[8]; uint4 q[2]; v16bf v; };

__device__ __forceinline__ unsigned int pk_bf16(float a, float b) {
  union { float f; unsigned int u; } x, y;
  x.f = a; y.f = b;
  unsigned int ha = (x.u + 0x7FFFu + ((x.u >> 16) & 1u)) >> 16;
  unsigned int hb = (y.u + 0x7FFFu + ((y.u >> 16) & 1u)) >> 16;
  return (ha & 0xFFFFu) | (hb << 16);
}
__device__ __forceinline__ unsigned short f2bf(float a) {
  union { float f; unsigned int u; } x; x.f = a;
  return (unsigned short)((x.u + 0x7FFFu + ((x.u >> 16) & 1u)) >> 16);
}
__device__ __forceinline__ float bf2f(unsigned short h) {
  union { unsigned int u; float f; } x; x.u = ((unsigned int)h) << 16;
  return x.f;
}
// tanh(x) = 1 - 2/(exp(2x)+1): v_exp_f32 + v_rcp_f32, saturates correctly at +-1
__device__ __forceinline__ float fast_tanh(float x) {
  float e = __expf(2.0f * x);
  return 1.0f - 2.0f * __builtin_amdgcn_rcpf(e + 1.0f);
}

__global__ __launch_bounds__(256, 1)
void attn_fused(const float* __restrict__ query,
                const float* __restrict__ keys,
                const int*   __restrict__ mask,
                const float* __restrict__ Wq,
                const float* __restrict__ bq,
                const float* __restrict__ Wk,
                const float* __restrict__ bk,
                const float* __restrict__ Ws,
                const float* __restrict__ bs,
                float* __restrict__ outp,
                float* __restrict__ woutp)
{
  __shared__ unsigned short keys_lds[SM * KP];    // 176,384 B : keys[b] in bf16 (padded)
  __shared__ unsigned short wk_lds[NH0 * KPW];    //  94,976 B : Wk n-half, transposed [n][k]
  __shared__ float qpw_s[HP];                     // qproj + bq + bk
  __shared__ float wsv_s[HP];                     // Ws (zero padded)
  __shared__ float scores_s[SM];
  __shared__ float red_s[16];

  const int tid  = threadIdx.x;
  const int lane = tid & 31;
  const int wave = tid >> 5;
  const int b    = blockIdx.x;

  // ---------- Phase 1: keys[b] (200x400 f32, HBM, read once) -> bf16 LDS ----------
  {
    const float4* kb = reinterpret_cast<const float4*>(keys + (size_t)b * SEQ * KD);
    for (int i = tid; i < SEQ * (KD / 4); i += 256) {
      int r = i / (KD / 4), c = i - r * (KD / 4);
      float4 v = kb[i];
      uint2 p; p.x = pk_bf16(v.x, v.y); p.y = pk_bf16(v.z, v.w);
      *reinterpret_cast<uint2*>(keys_lds + r * KP + c * 4) = p;
    }
    // zero pad cols [400, KP) for real rows
    for (int i = tid; i < SEQ * ((KP - KD) / 2); i += 256) {
      int r = i / ((KP - KD) / 2), c = i - r * ((KP - KD) / 2);
      *reinterpret_cast<unsigned int*>(keys_lds + r * KP + KD + c * 2) = 0u;
    }
    // zero pad rows [200, 208)
    for (int i = tid; i < (SM - SEQ) * KP / 2; i += 256)
      *reinterpret_cast<unsigned int*>(keys_lds + SEQ * KP + i * 2) = 0u;
  }

  // ---------- Phase 2: q_proj (+bq+bk), Ws staging, zero scores ----------
  if (tid < HP) {
    float acc = 0.f, wv = 0.f;
    if (tid < HD) {
      acc = bq[tid] + bk[tid];
      const float* q = query + (size_t)b * KD;
      #pragma unroll 4
      for (int k = 0; k < KD; ++k) acc = fmaf(q[k], Wq[k * HD + tid], acc);
      wv = Ws[tid];
    }
    qpw_s[tid] = acc;
    wsv_s[tid] = wv;
  }
  if (tid < SM) scores_s[tid] = 0.f;
  __syncthreads();

  // ---------- Phase 3: fused k_proj GEMM + score epilogue, two N-halves ----------
  for (int nh = 0; nh < 2; ++nh) {
    const int nbase = nh ? NH0 : 0;
    const int ncnt  = nh ? NH1 : NH0;

    // stage Wk[:, nbase:nbase+ncnt] transposed to bf16 LDS (L2-resident source)
    for (int i = tid; i < 13 * 32 * ncnt; i += 256) {
      int k = i / ncnt, j = i - k * ncnt;
      int n = nbase + j;
      float w = (k < KD && n < HD) ? Wk[k * HD + n] : 0.f;
      wk_lds[j * KPW + k] = f2bf(w);
    }
    __syncthreads();

    const int ntiles = 13 * (ncnt >> 4);
    const int mrow = lane & 15;
    const int lo   = (lane >> 4) << 3;   // 0 / 8 : half of A's K sub-offset (2*lo), M sub-offset for C
    for (int t = wave; t < ntiles; t += NWAVES) {
      const int mt = t % 13, nt = t / 13;
      const unsigned short* arow = keys_lds + (mt * 16 + mrow) * KP;
      const unsigned short* brow = wk_lds + (nt * 16 + mrow) * KPW + ((lane >> 4) << 4);
      v8f c = {0.f, 0.f, 0.f, 0.f, 0.f, 0.f, 0.f, 0.f};
      #pragma unroll 2
      for (int kt = 0; kt < 13; ++kt) {
        Frag a, bm;
        const int kb0 = kt * 32;
        // A 16x32 bf16: lanes 0-15 K 0-15, lanes 16-31 K 16-31, contiguous
        a.q[0] = *reinterpret_cast<const uint4*>(arow + kb0 + 2 * lo);
        a.q[1] = *reinterpret_cast<const uint4*>(arow + kb0 + 2 * lo + 8);
        // B 32x16 bf16: lane n holds K 0-15 (lanes 0-15) / 16-31 (lanes 16-31), contiguous
        bm.q[0] = *reinterpret_cast<const uint4*>(brow + kb0);
        bm.q[1] = *reinterpret_cast<const uint4*>(brow + kb0 + 8);
        c = __builtin_amdgcn_wmma_f32_16x16x32_bf16(false, a.v, false, bm.v,
                                                    (short)0, c, false, false);
      }
      // epilogue: scores[m] += sum_n tanh(qpw[n] + kproj[m][n]) * Ws[n]
      const int n = nbase + nt * 16 + mrow;
      const float qv = qpw_s[n], wv = wsv_s[n];
      #pragma unroll
      for (int r = 0; r < 8; ++r) {
        float tt = fast_tanh(qv + c[r]) * wv;
        tt += __shfl_xor(tt, 8, 16);
        tt += __shfl_xor(tt, 4, 16);
        tt += __shfl_xor(tt, 2, 16);
        tt += __shfl_xor(tt, 1, 16);
        if (mrow == 0) atomicAdd(&scores_s[mt * 16 + r + lo], tt);  // ds_add_f32
      }
    }
    __syncthreads();
  }

  // ---------- Phase 4: bias + mask + softmax over SEQ ----------
  const float bs0 = bs[0];
  float sc = -3.0e38f;
  if (tid < SEQ) {
    sc = scores_s[tid] + bs0;
    if (mask[(size_t)b * SEQ + tid] == 0) sc = -1.0e9f;
  }
  float mx = sc;
  mx = fmaxf(mx, __shfl_xor(mx, 16, 32));
  mx = fmaxf(mx, __shfl_xor(mx, 8, 32));
  mx = fmaxf(mx, __shfl_xor(mx, 4, 32));
  mx = fmaxf(mx, __shfl_xor(mx, 2, 32));
  mx = fmaxf(mx, __shfl_xor(mx, 1, 32));
  if (lane == 0) red_s[wave] = mx;
  __syncthreads();
  mx = red_s[0];
  #pragma unroll
  for (int i = 1; i < 8; ++i) mx = fmaxf(mx, red_s[i]);

  float e = 0.f;
  if (tid < SEQ) e = __expf(sc - mx);
  float sm = e;
  sm += __shfl_xor(sm, 16, 32);
  sm += __shfl_xor(sm, 8, 32);
  sm += __shfl_xor(sm, 4, 32);
  sm += __shfl_xor(sm, 2, 32);
  sm += __shfl_xor(sm, 1, 32);
  if (lane == 0) red_s[8 + wave] = sm;
  __syncthreads();
  sm = 0.f;
  #pragma unroll
  for (int i = 0; i < 8; ++i) sm += red_s[8 + i];
  const float inv = 1.0f / sm;
  if (tid < SEQ) {
    float w = e * inv;
    scores_s[tid] = w;
    woutp[(size_t)b * SEQ + tid] = w;   // weights output
  }
  __syncthreads();

  // ---------- Phase 5: output[b][k] = sum_s w[s] * keys[b][s][k] (all from LDS) ----------
  for (int k = tid; k < KD; k += 256) {
    float acc = 0.f;
    #pragma unroll 4
    for (int s = 0; s < SEQ; ++s)
      acc = fmaf(scores_s[s], bf2f(keys_lds[s * KP + k]), acc);
    outp[(size_t)b * KD + k] = acc;
  }
}

extern "C" void kernel_launch(void* const* d_in, const int* in_sizes, int n_in,
                              void* d_out, int out_size, void* d_ws, size_t ws_size,
                              hipStream_t stream) {
  (void)in_sizes; (void)n_in; (void)out_size; (void)d_ws; (void)ws_size;
  const float* query = (const float*)d_in[0];
  const float* keys  = (const float*)d_in[1];
  const int*   mask  = (const int*)d_in[2];
  const float* Wq    = (const float*)d_in[3];
  const float* bq    = (const float*)d_in[4];
  const float* Wk    = (const float*)d_in[5];
  const float* bk    = (const float*)d_in[6];
  const float* Ws    = (const float*)d_in[7];
  const float* bs    = (const float*)d_in[8];
  float* outp  = (float*)d_out;
  float* woutp = outp + (size_t)BATCH * KD;
  hipLaunchKernelGGL(attn_fused, dim3(BATCH), dim3(256), 0, stream,
                     query, keys, mask, Wq, bq, Wk, bk, Ws, bs, outp, woutp);
}
